// CNNBiaffine_29695403885099
// MI455X (gfx1250) — matrix operand order repaired and checked by reading, not compiled, not run on hardware
//
#include <hip/hip_runtime.h>
#include <hip/hip_bf16.h>
#include <math.h>

// ---------------------------------------------------------------------------
// CDNA5 / gfx1250 implementation. wave32, WMMA bf16 16x16x32.
// 32x32 register-blocked wave tiles on the dense GEMMs (16 FLOP/byte of
// operand traffic), LDS-staged conv window, flash attention with LDS-staged
// bf16 probability chunks feeding WMMA A-operands.
// ---------------------------------------------------------------------------

typedef __bf16 bf16_t;
typedef bf16_t v16bf __attribute__((ext_vector_type(16)));
typedef bf16_t v8bf  __attribute__((ext_vector_type(8)));
typedef float  v8f   __attribute__((ext_vector_type(8)));

#define BATCH 8
#define SEQ   2048
#define DM    768
#define NBS   (BATCH * SEQ)      // 16384 rows
#define FCIN  90

// ---- WMMA fragment loaders -------------------------------------------------
// A operand (16x32 bf16, row major source): lane r<16 holds row r, K {0..7,16..23}
// lane r>=16 holds row r-16, K {8..15,24..31}.  Caller pre-adjusts pointer by +8
// for hi lanes; chunks are then at +0 and +16.
__device__ __forceinline__ v16bf load_afrag(const bf16_t* p) {
  v8bf lo = *reinterpret_cast<const v8bf*>(p);
  v8bf hi = *reinterpret_cast<const v8bf*>(p + 16);
  return __builtin_shufflevector(lo, hi, 0,1,2,3,4,5,6,7,8,9,10,11,12,13,14,15);
}
// B operand (32x16 bf16, column-major source): lane n<16 holds col n, K {0..7,16..23}
// lanes 16..31 hold K {8..15,24..31} (caller pre-adjusts +8).
__device__ __forceinline__ v16bf load_bfrag(const bf16_t* p) {
  v8bf lo = *reinterpret_cast<const v8bf*>(p);
  v8bf hi = *reinterpret_cast<const v8bf*>(p + 16);
  return __builtin_shufflevector(lo, hi, 0,1,2,3,4,5,6,7,8,9,10,11,12,13,14,15);
}

__device__ __forceinline__ v8f wmma_bf16(v16bf a, v16bf b, v8f c) {
  return __builtin_amdgcn_wmma_f32_16x16x32_bf16(false, a, false, b, (short)0, c,
                                                 false, false);
}

// ---- 1. pack activations to bf16 (+ per-batch transpose of x2) -------------
__global__ __launch_bounds__(256) void k_pack_x(const float* __restrict__ x1,
                                                const float* __restrict__ x2,
                                                bf16_t* __restrict__ x1h,
                                                bf16_t* __restrict__ x2h,
                                                bf16_t* __restrict__ x2t) {
  size_t i = (size_t)blockIdx.x * 256 + threadIdx.x;
  if (i >= (size_t)NBS * DM) return;
  float a = x1[i];
  float c = x2[i];
  x1h[i] = (bf16_t)a;
  x2h[i] = (bf16_t)c;
  size_t d = i % DM, bt = i / DM;
  size_t t = bt & (SEQ - 1), b = bt >> 11;
  x2t[(b * DM + d) * SEQ + t] = (bf16_t)c;   // [b][d][t]
}

// ---- 2. pack weights: U^T and fcx_w to bf16 --------------------------------
__global__ __launch_bounds__(256) void k_pack_w(const float* __restrict__ U,
                                                const float* __restrict__ fcxw,
                                                bf16_t* __restrict__ Ut,
                                                bf16_t* __restrict__ fcxh) {
  int i = blockIdx.x * 256 + threadIdx.x;
  if (i < DM * 2 * DM) fcxh[i] = (bf16_t)fcxw[i];     // already [n][k]
  if (i < DM * DM) {
    int d = i / DM, e = i % DM;
    Ut[(size_t)e * DM + d] = (bf16_t)U[i];            // Ut[e][d] = U[d][e]
  }
}

// ---- 3. bias2[b,t] = x2[b,t,:] . bias_v ------------------------------------
__global__ __launch_bounds__(256) void k_bias2(const float* __restrict__ x2,
                                               const float* __restrict__ bias_v,
                                               float* __restrict__ bias2) {
  int row  = (blockIdx.x * 256 + threadIdx.x) >> 5;   // wave per row
  int lane = threadIdx.x & 31;
  const float* xr = x2 + (size_t)row * DM;
  float a = 0.f;
  for (int c = lane; c < DM; c += 32) a += xr[c] * bias_v[c];
  #pragma unroll
  for (int off = 16; off > 0; off >>= 1) a += __shfl_xor(a, off, 32);
  if (lane == 0) bias2[row] = a;
}

// ---- 4. conv(7/5/3) + relu + FC(90->768) + tanh  -> bf16 -------------------
// The 7-row input window (21.5 KB) is staged once into LDS; the 90 feature
// threads then stream weights from L2 and activations from LDS.
__global__ __launch_bounds__(128) void k_conv(const float* __restrict__ x2,
                                              const float* __restrict__ w7,
                                              const float* __restrict__ b7,
                                              const float* __restrict__ w5,
                                              const float* __restrict__ b5,
                                              const float* __restrict__ w3,
                                              const float* __restrict__ b3,
                                              const float* __restrict__ fcw,
                                              const float* __restrict__ fcb,
                                              bf16_t* __restrict__ xcnnh) {
  __shared__ float xs[7][DM];          // rows s-3 .. s+3, zero padded
  __shared__ float feat[FCIN];
  int bs = blockIdx.x;                 // one (b, s) position per block
  int b = bs >> 11, s = bs & (SEQ - 1);
  int t = threadIdx.x;
  for (int i = t; i < 7 * DM; i += 128) {
    int rr = i / DM, c = i - rr * DM;
    int si = s + rr - 3;
    xs[rr][c] = (si >= 0 && si < SEQ) ? x2[((size_t)b * SEQ + si) * DM + c] : 0.f;
  }
  __syncthreads();
  if (t < FCIN) {
    int k, o;
    const float *w, *bias;
    if (t < 40)      { k = 7; o = t;      w = w7; bias = b7; }
    else if (t < 70) { k = 5; o = t - 40; w = w5; bias = b5; }
    else             { k = 3; o = t - 70; w = w3; bias = b3; }
    float acc = bias[o];
    int h = k >> 1;
    for (int dk = 0; dk < k; ++dk) {
      const float* xr = xs[dk - h + 3];
      const float* wr = w + (size_t)o * DM * k + dk;   // stride k over channels
      float a = 0.f;
      for (int c = 0; c < DM; ++c) a += xr[c] * wr[(size_t)c * k];
      acc += a;
    }
    feat[t] = fmaxf(acc, 0.f);
  }
  __syncthreads();
  for (int n = t; n < DM; n += 128) {
    float a = fcb[n];
    const float* fw = fcw + (size_t)n * FCIN;
    #pragma unroll 10
    for (int j = 0; j < FCIN; ++j) a += feat[j] * fw[j];
    xcnnh[(size_t)bs * DM + n] = (bf16_t)tanhf(a);
  }
}

// ---- 5. Q = x1 @ U  (WMMA bf16, 32x32 wave tile, result bf16) --------------
__global__ __launch_bounds__(256) void k_gemm_q(const bf16_t* __restrict__ A,
                                                const bf16_t* __restrict__ Bcol,
                                                bf16_t* __restrict__ C) {
  const int lane = threadIdx.x & 31, w = threadIdx.x >> 5;
  const int r = lane & 15;
  const bool hi = lane >= 16;
  int tile = blockIdx.x * 8 + w;            // 512 mtiles * 24 ntiles
  int m0 = (tile / 24) * 32, n0 = (tile % 24) * 32;
  const bf16_t* a0 = A    + (size_t)(m0 + r) * DM + (hi ? 8 : 0);
  const bf16_t* a1 = a0 + (size_t)16 * DM;
  const bf16_t* b0 = Bcol + (size_t)(n0 + r) * DM + (hi ? 8 : 0);
  const bf16_t* b1 = b0 + (size_t)16 * DM;
  v8f c00 = {}, c01 = {}, c10 = {}, c11 = {};
  #pragma unroll 2
  for (int k0 = 0; k0 < DM; k0 += 32) {
    v16bf af0 = load_afrag(a0 + k0);
    v16bf af1 = load_afrag(a1 + k0);
    v16bf bf0 = load_bfrag(b0 + k0);
    v16bf bf1 = load_bfrag(b1 + k0);
    c00 = wmma_bf16(af0, bf0, c00);
    c01 = wmma_bf16(af0, bf1, c01);
    c10 = wmma_bf16(af1, bf0, c10);
    c11 = wmma_bf16(af1, bf1, c11);
  }
  int col0 = n0 + r, col1 = col0 + 16;
  #pragma unroll
  for (int j = 0; j < 8; ++j) {
    size_t row0 = m0 + j + (hi ? 8 : 0);
    size_t row1 = row0 + 16;
    C[row0 * DM + col0] = (bf16_t)c00[j];
    C[row0 * DM + col1] = (bf16_t)c01[j];
    C[row1 * DM + col0] = (bf16_t)c10[j];
    C[row1 * DM + col1] = (bf16_t)c11[j];
  }
}

// ---- 6. flash attention + tanh + residual ----------------------------------
// one workgroup (8 waves) per (b, 16-row s-tile); softmax over t in [0,2048)
__global__ __launch_bounds__(256) void k_attn(const bf16_t* __restrict__ qh,
                                              const bf16_t* __restrict__ kh,   // x2h
                                              const bf16_t* __restrict__ vt,   // x2t [b][d][t]
                                              const float*  __restrict__ bias2,
                                              const bf16_t* __restrict__ xcnnh,
                                              bf16_t* __restrict__ xresh) {
  __shared__ float  sS[8][16][17];      // per-wave score tile (padded)
  __shared__ float  lm[8][16], lsum[8][16];
  __shared__ float  gm[16], gs[16];
  __shared__ bf16_t P[16][128];         // prob chunk, bf16 A-operand
  const int lane = threadIdx.x & 31, w = threadIdx.x >> 5;
  const int r = lane & 15;
  const bool hi = lane >= 16;
  const int b = blockIdx.x >> 7;
  const int m0 = (blockIdx.x & 127) * 16;
  const bf16_t* qrow = qh + ((size_t)(b * SEQ + m0 + r)) * DM + (hi ? 8 : 0);
  const float*  bb   = bias2 + (size_t)b * SEQ;

  // ---- phase 1: row max / row sum over all 128 t-tiles (waves split tiles)
  float rm = -1e30f, rs = 0.f;
  for (int tt = w; tt < 128; tt += 8) {
    const bf16_t* bp = kh + ((size_t)(b * SEQ + tt * 16 + r)) * DM + (hi ? 8 : 0);
    v8f acc = {};
    #pragma unroll 4
    for (int k0 = 0; k0 < DM; k0 += 32)
      acc = wmma_bf16(load_afrag(qrow + k0), load_bfrag(bp + k0), acc);
    float bv = bb[tt * 16 + r];
    #pragma unroll
    for (int j = 0; j < 8; ++j) sS[w][j + (hi ? 8 : 0)][r] = acc[j] + bv;
    if (lane < 16) {                    // lane r reduces its row
      float tmax = -1e30f;
      #pragma unroll
      for (int c = 0; c < 16; ++c) tmax = fmaxf(tmax, sS[w][r][c]);
      float tsum = 0.f;
      #pragma unroll
      for (int c = 0; c < 16; ++c) tsum += __expf(sS[w][r][c] - tmax);
      float nm = fmaxf(rm, tmax);
      rs = rs * __expf(rm - nm) + tsum * __expf(tmax - nm);
      rm = nm;
    }
  }
  if (lane < 16) { lm[w][r] = rm; lsum[w][r] = rs; }
  __syncthreads();
  if (threadIdx.x < 16) {               // merge the 8 waves' partials
    float m = -1e30f, s = 0.f;
    #pragma unroll
    for (int ww = 0; ww < 8; ++ww) {
      float nm = fmaxf(m, lm[ww][threadIdx.x]);
      s = s * __expf(m - nm) + lsum[ww][threadIdx.x] * __expf(lm[ww][threadIdx.x] - nm);
      m = nm;
    }
    gm[threadIdx.x] = m;
    gs[threadIdx.x] = s;
  }
  __syncthreads();

  // ---- phase 2: O = P @ V, wave w owns output cols [w*96, w*96+96)
  v8f o[6];
  #pragma unroll
  for (int nt = 0; nt < 6; ++nt) o[nt] = (v8f){};
  for (int cc = 0; cc < 16; ++cc) {     // chunk = 128 t columns
    const int tt = cc * 8 + w;          // this wave's score tile in chunk
    const bf16_t* bp = kh + ((size_t)(b * SEQ + tt * 16 + r)) * DM + (hi ? 8 : 0);
    v8f acc = {};
    #pragma unroll 4
    for (int k0 = 0; k0 < DM; k0 += 32)
      acc = wmma_bf16(load_afrag(qrow + k0), load_bfrag(bp + k0), acc);
    float bv = bb[tt * 16 + r];
    __syncthreads();                    // prev chunk P fully consumed
    #pragma unroll
    for (int j = 0; j < 8; ++j) {
      int row = j + (hi ? 8 : 0);
      P[row][w * 16 + r] = (bf16_t)__expf(acc[j] + bv - gm[row]);
    }
    __syncthreads();                    // P chunk published
    const bf16_t* prow = &P[r][0] + (hi ? 8 : 0);
    #pragma unroll
    for (int nt = 0; nt < 6; ++nt) {
      int d = w * 96 + nt * 16 + r;
      const bf16_t* vp = vt + ((size_t)(b * DM + d)) * SEQ + cc * 128 + (hi ? 8 : 0);
      #pragma unroll
      for (int kk = 0; kk < 4; ++kk)    // 4 x K=32 covers 128 t
        o[nt] = wmma_bf16(load_afrag(prow + kk * 32), load_bfrag(vp + kk * 32), o[nt]);
    }
  }
  // ---- finalize: x = xcnn + tanh(O / rowsum)
  #pragma unroll
  for (int nt = 0; nt < 6; ++nt) {
    int d = w * 96 + nt * 16 + r;
    #pragma unroll
    for (int j = 0; j < 8; ++j) {
      int row = j + (hi ? 8 : 0);
      size_t idx = ((size_t)(b * SEQ + m0 + row)) * DM + d;
      float val = tanhf(o[nt][j] / gs[row]) + (float)xcnnh[idx];
      xresh[idx] = (bf16_t)val;
    }
  }
}

// ---- 7. out = [x1, x] @ fcx_w^T + fcx_b  (K = 1536, 32x32 tile, f32 out) ---
__global__ __launch_bounds__(256) void k_gemm_fcx(const bf16_t* __restrict__ X1h,
                                                  const bf16_t* __restrict__ Xresh,
                                                  const bf16_t* __restrict__ Bcol,
                                                  const float*  __restrict__ bias,
                                                  float* __restrict__ out) {
  const int lane = threadIdx.x & 31, w = threadIdx.x >> 5;
  const int r = lane & 15;
  const bool hi = lane >= 16;
  int tile = blockIdx.x * 8 + w;
  int m0 = (tile / 24) * 32, n0 = (tile % 24) * 32;
  const bf16_t* a0 = X1h   + (size_t)(m0 + r) * DM + (hi ? 8 : 0);
  const bf16_t* a1 = a0 + (size_t)16 * DM;
  const bf16_t* e0 = Xresh + (size_t)(m0 + r) * DM + (hi ? 8 : 0);
  const bf16_t* e1 = e0 + (size_t)16 * DM;
  const bf16_t* b0 = Bcol  + (size_t)(n0 + r) * (2 * DM) + (hi ? 8 : 0);
  const bf16_t* b1 = b0 + (size_t)16 * (2 * DM);
  v8f c00 = {}, c01 = {}, c10 = {}, c11 = {};
  #pragma unroll 2
  for (int k0 = 0; k0 < DM; k0 += 32) {      // K half 1: x1
    v16bf af0 = load_afrag(a0 + k0);
    v16bf af1 = load_afrag(a1 + k0);
    v16bf bf0 = load_bfrag(b0 + k0);
    v16bf bf1 = load_bfrag(b1 + k0);
    c00 = wmma_bf16(af0, bf0, c00);
    c01 = wmma_bf16(af0, bf1, c01);
    c10 = wmma_bf16(af1, bf0, c10);
    c11 = wmma_bf16(af1, bf1, c11);
  }
  #pragma unroll 2
  for (int k0 = 0; k0 < DM; k0 += 32) {      // K half 2: x (attn+cnn result)
    v16bf af0 = load_afrag(e0 + k0);
    v16bf af1 = load_afrag(e1 + k0);
    v16bf bf0 = load_bfrag(b0 + DM + k0);
    v16bf bf1 = load_bfrag(b1 + DM + k0);
    c00 = wmma_bf16(af0, bf0, c00);
    c01 = wmma_bf16(af0, bf1, c01);
    c10 = wmma_bf16(af1, bf0, c10);
    c11 = wmma_bf16(af1, bf1, c11);
  }
  int col0 = n0 + r, col1 = col0 + 16;
  float bv0 = bias[col0], bv1 = bias[col1];
  #pragma unroll
  for (int j = 0; j < 8; ++j) {
    size_t row0 = m0 + j + (hi ? 8 : 0);
    size_t row1 = row0 + 16;
    out[row0 * DM + col0] = c00[j] + bv0;
    out[row0 * DM + col1] = c01[j] + bv1;
    out[row1 * DM + col0] = c10[j] + bv0;
    out[row1 * DM + col1] = c11[j] + bv1;
  }
}

// ---------------------------------------------------------------------------
extern "C" void kernel_launch(void* const* d_in, const int* in_sizes, int n_in,
                              void* d_out, int out_size, void* d_ws, size_t ws_size,
                              hipStream_t stream) {
  const float* x1     = (const float*)d_in[0];
  const float* x2     = (const float*)d_in[1];
  const float* w7     = (const float*)d_in[2];
  const float* b7     = (const float*)d_in[3];
  const float* w5     = (const float*)d_in[4];
  const float* b5     = (const float*)d_in[5];
  const float* w3     = (const float*)d_in[6];
  const float* b3     = (const float*)d_in[7];
  const float* U      = (const float*)d_in[8];
  const float* bias_v = (const float*)d_in[9];
  const float* fc_w   = (const float*)d_in[10];
  const float* fc_b   = (const float*)d_in[11];
  const float* fcx_w  = (const float*)d_in[12];
  const float* fcx_b  = (const float*)d_in[13];
  float* out = (float*)d_out;

  // workspace arena (~150 MB)
  char* p = (char*)d_ws;
  auto alloc = [&](size_t bytes) -> char* {
    char* ret = p;
    p += (bytes + 255) & ~(size_t)255;
    return ret;
  };
  bf16_t* x1h   = (bf16_t*)alloc((size_t)NBS * DM * 2);
  bf16_t* x2h   = (bf16_t*)alloc((size_t)NBS * DM * 2);
  bf16_t* x2t   = (bf16_t*)alloc((size_t)NBS * DM * 2);
  bf16_t* qh    = (bf16_t*)alloc((size_t)NBS * DM * 2);
  bf16_t* xcnnh = (bf16_t*)alloc((size_t)NBS * DM * 2);
  bf16_t* xresh = (bf16_t*)alloc((size_t)NBS * DM * 2);
  float*  bias2 = (float*)alloc((size_t)NBS * 4);
  bf16_t* Ut    = (bf16_t*)alloc((size_t)DM * DM * 2);
  bf16_t* fcxh  = (bf16_t*)alloc((size_t)DM * 2 * DM * 2);
  (void)ws_size; (void)in_sizes; (void)n_in; (void)out_size;

  k_pack_x<<<(NBS * DM + 255) / 256, 256, 0, stream>>>(x1, x2, x1h, x2h, x2t);
  k_pack_w<<<(DM * 2 * DM + 255) / 256, 256, 0, stream>>>(U, fcx_w, Ut, fcxh);
  k_bias2<<<NBS / 8, 256, 0, stream>>>(x2, bias_v, bias2);
  k_conv<<<NBS, 128, 0, stream>>>(x2, w7, b7, w5, b5, w3, b3, fc_w, fc_b, xcnnh);
  k_gemm_q<<<(NBS / 32) * (DM / 32) / 8, 256, 0, stream>>>(x1h, Ut, qh);
  k_attn<<<BATCH * (SEQ / 16), 256, 0, stream>>>(qh, x2h, x2t, bias2, xcnnh, xresh);
  k_gemm_fcx<<<(NBS / 32) * (DM / 32) / 8, 256, 0, stream>>>(x1h, xresh, fcxh,
                                                             fcx_b, out);
}
